// SelfAttention2D_70927089926205
// MI455X (gfx1250) — matrix operand order repaired and checked by reading, not compiled, not run on hardware
//
#include <hip/hip_runtime.h>

typedef __bf16 bf16_t;
typedef __attribute__((ext_vector_type(16))) __bf16 v16bf;
typedef __attribute__((ext_vector_type(8)))  __bf16 v8bf;
typedef __attribute__((ext_vector_type(8)))  float  v8f;
typedef __attribute__((ext_vector_type(4)))  unsigned int u32x4;
typedef __attribute__((ext_vector_type(8)))  int i32x8;
typedef __attribute__((ext_vector_type(4)))  int i32x4;

// ---------------------------------------------------------------------------
// WMMA helper: D = A(16x32 bf16) * B(32x16 bf16) + C(16x16 f32)
// ---------------------------------------------------------------------------
__device__ __forceinline__ v8f wmma_bf16(v16bf a, v16bf b, v8f c) {
  return __builtin_amdgcn_wmma_f32_16x16x32_bf16(
      /*neg_a=*/false, a, /*neg_b=*/false, b,
      /*c_mod=*/(short)0, c, /*reuse_a=*/false, /*reuse_b=*/false);
}

// A fragment (16x32, bf16) from row-major [*, ld] at (row0, k0).
// ISA: lanes 0-15 row=lane, K={0..7,16..23}; lanes 16-31 K={8..15,24..31}.
__device__ __forceinline__ v16bf load_A_frag(const bf16_t* base, int ld,
                                             int row0, int k0) {
  int lane = threadIdx.x & 31;
  int r    = lane & 15;
  int ksel = (lane >> 4) * 8;                 // 0 or 8
  const bf16_t* p = base + (size_t)(row0 + r) * ld + k0 + ksel;
  v8bf lo = *(const v8bf*)(p);
  v8bf hi = *(const v8bf*)(p + 16);
  v16bf a;
#pragma unroll
  for (int i = 0; i < 8; ++i) { a[i] = lo[i]; a[8 + i] = hi[i]; }
  return a;
}

// B fragment (32x16, bf16) where B[k,n] = W[n0+n, k0+k] (W row-major [*, ld]).
// ISA: lane = column n; lanes 0-15 K={0..7,16..23}; lanes 16-31 K={8..15,24..31}.
__device__ __forceinline__ v16bf load_B_frag(const bf16_t* base, int ld,
                                             int n0, int k0) {
  int lane = threadIdx.x & 31;
  int n    = lane & 15;
  int ksel = (lane >> 4) * 8;                 // 0 or 8
  const bf16_t* p = base + (size_t)(n0 + n) * ld + k0 + ksel;
  v8bf lo = *(const v8bf*)(p);
  v8bf hi = *(const v8bf*)(p + 16);
  v16bf b;
#pragma unroll
  for (int i = 0; i < 8; ++i) { b[i] = lo[i]; b[8 + i] = hi[i]; }
  return b;
}

// ---------------------------------------------------------------------------
// TDM: async 2D tile load global -> LDS (tensor_load_to_lds, TENSORcnt).
// D# per CDNA5 ISA: group0 {count=1, lds_addr, global_addr,
// type=2}; group1 {data_size=2B, tensor_dim0/1, tile_dim0/1, dim0_stride}.
// ---------------------------------------------------------------------------
__device__ __forceinline__ void tdm_load_2d(unsigned lds_off, const void* gptr,
                                            unsigned dim0, unsigned dim1,
                                            unsigned tile0, unsigned tile1,
                                            unsigned stride0) {
  unsigned long long ga = (unsigned long long)(uintptr_t)gptr;
  u32x4 g0 = { 1u,                                   // count=1, no gather
               lds_off,                              // LDS byte address
               (unsigned)ga,                         // global_addr[31:0]
               (unsigned)((ga >> 32) & 0x01FFFFFFu) | (2u << 30) }; // [56:32]|type=2
  i32x8 g1 = { (int)0x00010000u,                     // data_size=1 -> 2 bytes
               (int)((dim0 & 0xFFFFu) << 16),        // tensor_dim0[15:0]
               (int)((dim0 >> 16) | ((dim1 & 0xFFFFu) << 16)),
               (int)((dim1 >> 16) | (tile0 << 16)),  // tile_dim0
               (int)tile1,                           // tile_dim1 (tile_dim2=0)
               (int)stride0,                         // tensor_dim0_stride[31:0]
               0, 0 };
  i32x4 z4 = { 0, 0, 0, 0 };
#if __clang_major__ >= 23
  i32x8 z8 = { 0, 0, 0, 0, 0, 0, 0, 0 };
  __builtin_amdgcn_tensor_load_to_lds(g0, g1, z4, z4, z8, 0);
#else
  __builtin_amdgcn_tensor_load_to_lds(g0, g1, z4, z4, 0);
#endif
}

// ---------------------------------------------------------------------------
// fp32 -> bf16 conversion
// ---------------------------------------------------------------------------
__global__ void f32_to_bf16_kernel(const float* __restrict__ src,
                                   bf16_t* __restrict__ dst, int n) {
  int i = blockIdx.x * blockDim.x + threadIdx.x;
  if (i < n) dst[i] = (bf16_t)src[i];
}

// ---------------------------------------------------------------------------
// QKV GEMM: qkv[m,n] = sum_k x[m,k]*qkv_w[n,k] + qkv_b[n]
// M=2048, N=2304, K=768. One wave per 16x64 tile (A fragment reused 4x).
// Scatter: q,k -> [bh][1024][64] ; v -> transposed [bh][64][1024]
// ---------------------------------------------------------------------------
__global__ __launch_bounds__(256)
void qkv_gemm_kernel(const bf16_t* __restrict__ Xb, const bf16_t* __restrict__ Wb,
                     const float* __restrict__ bias,
                     bf16_t* __restrict__ qb, bf16_t* __restrict__ kb,
                     bf16_t* __restrict__ vTb) {
  int wave = threadIdx.x >> 5;
  int tile = blockIdx.x * 8 + wave;           // 4608 tiles exactly
  const int TN = 2304 / 64;                   // 36
  int tm = tile / TN, tn = tile % TN;
  int row0 = tm * 16, n0 = tn * 64;

  v8f acc[4] = {};
  for (int k0 = 0; k0 < 768; k0 += 32) {
    v16bf a = load_A_frag(Xb, 768, row0, k0);
    if (k0 + 32 < 768)                        // stream-prefetch next A chunk
      __builtin_prefetch(Xb + (size_t)(row0 + (threadIdx.x & 15)) * 768 + k0 + 32, 0, 0);
#pragma unroll
    for (int s = 0; s < 4; ++s)
      acc[s] = wmma_bf16(a, load_B_frag(Wb, 768, n0 + s * 16, k0), acc[s]);
  }

  int lane   = threadIdx.x & 31;
  int rowadd = (lane >> 4) * 8;
#pragma unroll
  for (int s = 0; s < 4; ++s) {
    int col   = n0 + s * 16 + (lane & 15);
    float bv  = bias[col];
    int which = col / 768;                    // 0=q 1=k 2=v
    int c     = col % 768;
    int h = c >> 6, d = c & 63;
#pragma unroll
    for (int r = 0; r < 8; ++r) {
      int m  = row0 + r + rowadd;
      int b_ = m >> 10, nq = m & 1023;
      bf16_t o = (bf16_t)(acc[s][r] + bv);
      size_t bh = (size_t)(b_ * 12 + h);
      if (which == 0)      qb [(bh * 1024 + nq) * 64 + d] = o;
      else if (which == 1) kb [(bh * 1024 + nq) * 64 + d] = o;
      else                 vTb[(bh * 64 + d) * 1024 + nq] = o;
    }
  }
}

// ---------------------------------------------------------------------------
// Decomposed rel-pos bias: rel_h[bh][q][k] = sum_c q[bh][q][c]*rel_pos_h[hq-k+31][c]
// ---------------------------------------------------------------------------
__global__ void rel_bias_kernel(const bf16_t* __restrict__ qb,
                                const float* __restrict__ rph,
                                const float* __restrict__ rpw,
                                float* __restrict__ rel_h,
                                float* __restrict__ rel_w) {
  const int PER = 2 * 12 * 1024 * 32;         // 786432
  int idx = blockIdx.x * blockDim.x + threadIdx.x;
  if (idx >= 2 * PER) return;
  int which = idx / PER;
  int t     = idx - which * PER;
  int k  = t & 31;
  int q  = (t >> 5) & 1023;
  int bh = t >> 15;
  int hq = q >> 5, wq = q & 31;
  const bf16_t* qrow = qb + ((size_t)bh * 1024 + q) * 64;
  const float*  tab  = which ? (rpw + (size_t)(wq - k + 31) * 64)
                             : (rph + (size_t)(hq - k + 31) * 64);
  float s = 0.f;
#pragma unroll 8
  for (int c = 0; c < 64; ++c) s += (float)qrow[c] * tab[c];
  (which ? rel_w : rel_h)[t] = s;
}

// ---------------------------------------------------------------------------
// Fused flash attention. Workgroup = (b,h, 64-row q block), 4 waves x 16 rows.
// K/V tiles (32 keys) are TDM-staged into double-buffered LDS and shared by
// all 4 waves; wave 0 issues tensor_load_to_lds for tile i+1, waits
// s_wait_tensorcnt for tile i, block barrier publishes the buffer.
// ---------------------------------------------------------------------------
__global__ __launch_bounds__(128)
void attn_kernel(const bf16_t* __restrict__ qb, const bf16_t* __restrict__ kb,
                 const bf16_t* __restrict__ vTb,
                 const float* __restrict__ rel_h, const float* __restrict__ rel_w,
                 bf16_t* __restrict__ attn_out) {
  // LDS map (single block => base offset 0 for TDM lds_addr):
  //   [0,8K)   K tiles  : 2 x [32 keys][64 ch]  bf16
  //   [8K,16K) V tiles  : 2 x [64 d][32 keys]   bf16
  //   [16K,20K) P tiles : 4 waves x [16][32]    bf16
  __shared__ __align__(16) unsigned char smem[20480];

  int bh   = blockIdx.x >> 4;                 // 0..23  (b*12+h)
  int qblk = blockIdx.x & 15;
  int h    = bh % 12, b_ = bh / 12;
  int wave = threadIdx.x >> 5;
  int q0   = qblk * 64 + wave * 16;

  const bf16_t* Q  = qb  + (size_t)bh * 1024 * 64;
  const bf16_t* K  = kb  + (size_t)bh * 1024 * 64;
  const bf16_t* VT = vTb + (size_t)bh * 64 * 1024;
  const float*  RH = rel_h + (size_t)bh * 1024 * 32;
  const float*  RW = rel_w + (size_t)bh * 1024 * 32;

  int lane   = threadIdx.x & 31;
  int rowadd = (lane >> 4) * 8;
  int cn     = lane & 15;

  bf16_t* Pw = (bf16_t*)(smem + 16384 + wave * 1024);   // this wave's P [16][32]

  v16bf qa0 = load_A_frag(Q, 64, q0, 0);
  v16bf qa1 = load_A_frag(Q, 64, q0, 32);

  float mrow[8], lrow[8];
  v8f acc[4] = {};
#pragma unroll
  for (int r = 0; r < 8; ++r) { mrow[r] = -1e30f; lrow[r] = 0.f; }

  // Prologue: stage tile 0 into buffer 0.
  if (wave == 0) {
    tdm_load_2d(0u,        K,  64u, 1024u, 64u, 32u, 64u);     // K: 32x64
    tdm_load_2d(8192u,     VT, 1024u, 64u, 32u, 64u, 1024u);   // V: 64x32
  }

  for (int kt = 0; kt < 1024; kt += 32) {
    unsigned buf = (unsigned)(kt >> 5) & 1u;
    if (wave == 0) {
      if (kt + 32 < 1024) {                   // issue next tile into buf^1
        unsigned nb = buf ^ 1u;
        tdm_load_2d(nb * 4096u,        K  + (size_t)(kt + 32) * 64,
                    64u, 1024u, 64u, 32u, 64u);
        tdm_load_2d(8192u + nb * 4096u, VT + (size_t)(kt + 32),
                    1024u, 64u, 32u, 64u, 1024u);
        __builtin_amdgcn_s_wait_tensorcnt((short)2);   // current tile done
      } else {
        __builtin_amdgcn_s_wait_tensorcnt((short)0);
      }
    }
    __syncthreads();                          // buffer `buf` visible to all

    const bf16_t* Kbuf = (const bf16_t*)(smem + buf * 4096);          // [32][64]
    const bf16_t* Vbuf = (const bf16_t*)(smem + 8192 + buf * 4096);   // [64][32]

    // ---- scores for keys kt..kt+31 (two 16-col tiles) from LDS K ----
    v8f s0 = {}; v8f s1 = {};
    s0 = wmma_bf16(qa0, load_B_frag(Kbuf, 64, 0,  0),  s0);
    s0 = wmma_bf16(qa1, load_B_frag(Kbuf, 64, 0,  32), s0);
    s1 = wmma_bf16(qa0, load_B_frag(Kbuf, 64, 16, 0),  s1);
    s1 = wmma_bf16(qa1, load_B_frag(Kbuf, 64, 16, 32), s1);

    int kn0 = kt + cn, kn1 = kt + 16 + cn;
    int hk0 = kn0 >> 5, wk0 = kn0 & 31;
    int hk1 = kn1 >> 5, wk1 = kn1 & 31;
#pragma unroll
    for (int r = 0; r < 8; ++r) {
      int qr = q0 + r + rowadd;
      s0[r] = s0[r] * 0.125f + RH[qr * 32 + hk0] + RW[qr * 32 + wk0];
      s1[r] = s1[r] * 0.125f + RH[qr * 32 + hk1] + RW[qr * 32 + wk1];
    }

    // ---- online softmax (row reductions across the 16-lane group) ----
    float alpha[8];
#pragma unroll
    for (int r = 0; r < 8; ++r) {
      float mx = fmaxf(s0[r], s1[r]);
#pragma unroll
      for (int off = 8; off >= 1; off >>= 1)
        mx = fmaxf(mx, __shfl_xor(mx, off, 32));
      float mnew = fmaxf(mrow[r], mx);
      alpha[r]   = __expf(mrow[r] - mnew);
      mrow[r]    = mnew;
      float p0 = __expf(s0[r] - mnew);
      float p1 = __expf(s1[r] - mnew);
      s0[r] = p0; s1[r] = p1;
      float rs = p0 + p1;
#pragma unroll
      for (int off = 8; off >= 1; off >>= 1)
        rs += __shfl_xor(rs, off, 32);
      lrow[r] = lrow[r] * alpha[r] + rs;
    }

#pragma unroll
    for (int f = 0; f < 4; ++f)
#pragma unroll
      for (int r = 0; r < 8; ++r) acc[f][r] *= alpha[r];

    // ---- C-layout -> A-layout reshape of P through this wave's LDS ----
#pragma unroll
    for (int r = 0; r < 8; ++r) {
      Pw[(r + rowadd) * 32 + cn]      = (bf16_t)s0[r];
      Pw[(r + rowadd) * 32 + cn + 16] = (bf16_t)s1[r];
    }
    __syncthreads();
    v16bf pa;
    {
      int rr = lane & 15, ksel = (lane >> 4) * 8;
      const bf16_t* p = Pw + rr * 32 + ksel;
      v8bf lo = *(const v8bf*)p;
      v8bf hi = *(const v8bf*)(p + 16);
#pragma unroll
      for (int i = 0; i < 8; ++i) { pa[i] = lo[i]; pa[8 + i] = hi[i]; }
    }

    // ---- PV: acc[f] += P(16x32) . V-tile columns f*16.. from LDS ----
#pragma unroll
    for (int f = 0; f < 4; ++f)
      acc[f] = wmma_bf16(pa, load_B_frag(Vbuf, 32, f * 16, 0), acc[f]);

    __syncthreads();    // all waves done with `buf` before next-iter TDM reuse
  }

  // ---- epilogue: normalize, write bf16 [B,N,768] ----
#pragma unroll
  for (int f = 0; f < 4; ++f)
#pragma unroll
    for (int r = 0; r < 8; ++r) {
      int m = q0 + r + rowadd;
      int d = f * 16 + cn;
      attn_out[((size_t)b_ * 1024 + m) * 768 + h * 64 + d] =
          (bf16_t)(acc[f][r] / lrow[r]);
    }
}

// ---------------------------------------------------------------------------
// Proj GEMM: out[m,n] = sum_k a[m,k]*proj_w[n,k] + proj_b[n]  (fp32 out)
// M=2048, N=768, K=768. One wave per 16x64 tile.
// ---------------------------------------------------------------------------
__global__ __launch_bounds__(256)
void proj_gemm_kernel(const bf16_t* __restrict__ Ab, const bf16_t* __restrict__ Wb,
                      const float* __restrict__ bias, float* __restrict__ out) {
  int wave = threadIdx.x >> 5;
  int tile = blockIdx.x * 8 + wave;           // 1536 tiles exactly
  const int TN = 768 / 64;                    // 12
  int tm = tile / TN, tn = tile % TN;
  int row0 = tm * 16, n0 = tn * 64;

  v8f acc[4] = {};
  for (int k0 = 0; k0 < 768; k0 += 32) {
    v16bf a = load_A_frag(Ab, 768, row0, k0);
    if (k0 + 32 < 768)
      __builtin_prefetch(Ab + (size_t)(row0 + (threadIdx.x & 15)) * 768 + k0 + 32, 0, 0);
#pragma unroll
    for (int s = 0; s < 4; ++s)
      acc[s] = wmma_bf16(a, load_B_frag(Wb, 768, n0 + s * 16, k0), acc[s]);
  }

  int lane   = threadIdx.x & 31;
  int rowadd = (lane >> 4) * 8;
#pragma unroll
  for (int s = 0; s < 4; ++s) {
    int col  = n0 + s * 16 + (lane & 15);
    float bv = bias[col];
#pragma unroll
    for (int r = 0; r < 8; ++r)
      out[(size_t)(row0 + r + rowadd) * 768 + col] = acc[s][r] + bv;
  }
}

// ---------------------------------------------------------------------------
extern "C" void kernel_launch(void* const* d_in, const int* in_sizes, int n_in,
                              void* d_out, int out_size, void* d_ws, size_t ws_size,
                              hipStream_t stream) {
  const float* x         = (const float*)d_in[0];
  const float* qkv_w     = (const float*)d_in[1];
  const float* qkv_b     = (const float*)d_in[2];
  const float* proj_w    = (const float*)d_in[3];
  const float* proj_b    = (const float*)d_in[4];
  const float* rel_pos_h = (const float*)d_in[5];
  const float* rel_pos_w = (const float*)d_in[6];
  // d_in[7]=x_h, d_in[8]=x_w : fixed at 32 for this problem instance.

  char* ws = (char*)d_ws;
  size_t off = 0;
  auto alloc = [&](size_t bytes) -> char* {
    char* p = ws + off;
    off += (bytes + 255) & ~(size_t)255;
    return p;
  };
  bf16_t* x_bf  = (bf16_t*)alloc((size_t)2048 * 768 * 2);
  bf16_t* wq_bf = (bf16_t*)alloc((size_t)2304 * 768 * 2);
  bf16_t* wp_bf = (bf16_t*)alloc((size_t)768 * 768 * 2);
  bf16_t* q_bf  = (bf16_t*)alloc((size_t)24 * 1024 * 64 * 2);
  bf16_t* k_bf  = (bf16_t*)alloc((size_t)24 * 1024 * 64 * 2);
  bf16_t* vT_bf = (bf16_t*)alloc((size_t)24 * 64 * 1024 * 2);
  float*  rel_h = (float*)alloc((size_t)24 * 1024 * 32 * 4);
  float*  rel_w = (float*)alloc((size_t)24 * 1024 * 32 * 4);
  bf16_t* a_bf  = (bf16_t*)alloc((size_t)2048 * 768 * 2);

  f32_to_bf16_kernel<<<(2048 * 768 + 255) / 256, 256, 0, stream>>>(x, x_bf, 2048 * 768);
  f32_to_bf16_kernel<<<(2304 * 768 + 255) / 256, 256, 0, stream>>>(qkv_w, wq_bf, 2304 * 768);
  f32_to_bf16_kernel<<<(768 * 768 + 255) / 256, 256, 0, stream>>>(proj_w, wp_bf, 768 * 768);

  qkv_gemm_kernel<<<576, 256, 0, stream>>>(x_bf, wq_bf, qkv_b, q_bf, k_bf, vT_bf);
  rel_bias_kernel<<<(2 * 786432 + 255) / 256, 256, 0, stream>>>(q_bf, rel_pos_h,
                                                                rel_pos_w, rel_h, rel_w);
  attn_kernel<<<384, 128, 0, stream>>>(q_bf, k_bf, vT_bf, rel_h, rel_w, a_bf);
  proj_gemm_kernel<<<192, 256, 0, stream>>>(a_bf, wp_bf, proj_b, (float*)d_out);
}
